// EWNet_13554916786250
// MI455X (gfx1250) — matrix operand order repaired and checked by reading, not compiled, not run on hardware
//
#include <hip/hip_runtime.h>
#include <hip/hip_bf16.h>

#define N_NODES 6144
#define EMB 256
#define HID 128
#define SIGMA_INV (1.0f / 100.0f)
#define T_DELETE 0.1f

typedef __attribute__((ext_vector_type(16))) __bf16        v16bf;
typedef __attribute__((ext_vector_type(8)))  float         v8f;
typedef __attribute__((ext_vector_type(8)))  unsigned int  v8u;

union Frag { v8u u; v16bf h; };

__device__ __forceinline__ unsigned short f2bf(float f) {
    union { float f; unsigned u; } v; v.f = f;
    unsigned r = v.u + 0x7FFFu + ((v.u >> 16) & 1u);   // round-to-nearest-even
    return (unsigned short)(r >> 16);
}
__device__ __forceinline__ unsigned pack2(float lo, float hi) {
    return (unsigned)f2bf(lo) | ((unsigned)f2bf(hi) << 16);
}
__device__ __forceinline__ float sigmoidf(float x) {
    return 1.0f / (1.0f + expf(-x));
}

// ---------------------------------------------------------------------------
// One-pass f32 -> packed bf16 conversion (hoists ALL conversion work out of
// the GEMM hot loop; converted operands live in d_ws and are L2-resident).
// ---------------------------------------------------------------------------
__global__ __launch_bounds__(256) void cvt_bf16_kernel(
    const float* __restrict__ src, unsigned* __restrict__ dst, int nPairs)
{
    int idx = blockIdx.x * blockDim.x + threadIdx.x;
    const int stride = gridDim.x * blockDim.x;
    for (; idx < nPairs; idx += stride) {
        float2 f = ((const float2*)src)[idx];
        dst[idx] = pack2(f.x, f.y);
    }
}

// ---------------------------------------------------------------------------
// Fused 2-layer MLP, both branches (blockIdx.y selects link / aa weights).
// 256 threads = 8 wave32; each wave computes a 16x128 latent tile with
// v_wmma_f32_16x16x32_bf16. All operands pre-converted bf16: each A fragment
// is two 128-bit loads (global for emb, ds for the LDS-staged hidden tile); W is 16 strided loads.
// ---------------------------------------------------------------------------
__global__ __launch_bounds__(256) void mlp_wmma_kernel(
    const unsigned short* __restrict__ embB,
    const unsigned short* __restrict__ W1LB, const float* __restrict__ b1L,
    const unsigned short* __restrict__ W2LB, const float* __restrict__ b2L,
    const unsigned short* __restrict__ W1AB, const float* __restrict__ b1A,
    const unsigned short* __restrict__ W2AB, const float* __restrict__ b2A,
    float* __restrict__ latL, float* __restrict__ latA)
{
    __shared__ unsigned short sH[8 * 16 * HID];   // 8 waves x 16x128 bf16 = 32KB

    const int branch = blockIdx.y;
    const unsigned short* W1 = branch ? W1AB : W1LB;
    const float*          b1 = branch ? b1A  : b1L;
    const unsigned short* W2 = branch ? W2AB : W2LB;
    const float*          b2 = branch ? b2A  : b2L;
    float*               lat = branch ? latA : latL;

    const int wave = threadIdx.x >> 5;
    const int lane = threadIdx.x & 31;
    const int m    = lane & 15;       // A row / B,C,D column within 16-tile
    const int half = lane >> 4;
    const int r0   = blockIdx.x * 128 + wave * 16;

    // ---------------- GEMM1: [16x256] x [256x128], ReLU -------------------
    v8f acc[8] = {};
    const unsigned short* aRow = embB + (size_t)(r0 + m) * EMB;
    for (int ks = 0; ks < EMB / 32; ++ks) {
        // A fragment 16x32 bf16: lane M=lane&15; two contiguous 16B half-frags
        Frag a;
        uint4 t0 = *(const uint4*)(aRow + ks * 32 + half * 8);
        uint4 t1 = *(const uint4*)(aRow + ks * 32 + 16 + half * 8);
        a.u[0] = t0.x; a.u[1] = t0.y; a.u[2] = t0.z; a.u[3] = t0.w;
        a.u[4] = t1.x; a.u[5] = t1.y; a.u[6] = t1.z; a.u[7] = t1.w;
#pragma unroll
        for (int nt = 0; nt < 8; ++nt) {
            // B fragment 32x16 bf16: lane N=lane&15, element s = K (table order); 16 strided bf16
            Frag b;
            const unsigned short* bcol = W1 + (size_t)(ks * 32) * HID + nt * 16 + m;
#pragma unroll
            for (int s = 0; s < 16; ++s) {
                const int kk = (s < 8) ? (8 * half + s) : (16 + 8 * half + s - 8);
                unsigned short v = bcol[(size_t)kk * HID];
                b.h[s] = __builtin_bit_cast(__bf16, v);
            }
            acc[nt] = __builtin_amdgcn_wmma_f32_16x16x32_bf16(
                false, a.h, false, b.h, (short)0, acc[nt], false, false);
        }
    }

    // bias + ReLU -> bf16 H tile in LDS (C/D layout: VGPR g -> row g + 8*half)
    unsigned short* hTile = sH + wave * 16 * HID;
#pragma unroll
    for (int nt = 0; nt < 8; ++nt) {
        const float bias = b1[nt * 16 + m];
#pragma unroll
        for (int g = 0; g < 8; ++g) {
            float hv = fmaxf(acc[nt][g] + bias, 0.0f);
            hTile[(g + 8 * half) * HID + nt * 16 + m] = f2bf(hv);
        }
    }
    __syncthreads();

    // ---------------- GEMM2: [16x128] x [128x128] -------------------------
    v8f acc2[8] = {};
    const unsigned short* hRow = hTile + m * HID;
#pragma unroll
    for (int ks = 0; ks < HID / 32; ++ks) {
        Frag a;
        uint4 t0 = *(const uint4*)(hRow + ks * 32 + half * 8);       // ds_load_b128
        uint4 t1 = *(const uint4*)(hRow + ks * 32 + 16 + half * 8);  // ds_load_b128
        a.u[0] = t0.x; a.u[1] = t0.y; a.u[2] = t0.z; a.u[3] = t0.w;
        a.u[4] = t1.x; a.u[5] = t1.y; a.u[6] = t1.z; a.u[7] = t1.w;
#pragma unroll
        for (int nt = 0; nt < 8; ++nt) {
            Frag b;
            const unsigned short* bcol = W2 + (size_t)(ks * 32) * HID + nt * 16 + m;
#pragma unroll
            for (int s = 0; s < 16; ++s) {
                const int kk = (s < 8) ? (8 * half + s) : (16 + 8 * half + s - 8);
                unsigned short v = bcol[(size_t)kk * HID];
                b.h[s] = __builtin_bit_cast(__bf16, v);
            }
            acc2[nt] = __builtin_amdgcn_wmma_f32_16x16x32_bf16(
                false, a.h, false, b.h, (short)0, acc2[nt], false, false);
        }
    }

    // bias + store fp32 latent
#pragma unroll
    for (int nt = 0; nt < 8; ++nt) {
        const float bias = b2[nt * 16 + m];
#pragma unroll
        for (int g = 0; g < 8; ++g) {
            lat[(size_t)(r0 + g + 8 * half) * HID + nt * 16 + m] = acc2[nt][g] + bias;
        }
    }
}

// ---------------------------------------------------------------------------
// Edge scoring. One wave32 per edge (grid-strided). Two 128-dim dot products
// via float4 lane loads + shfl_xor reduction. Latents (6.3MB) are L2-resident
// on the 192MB L2, so gathers are L2-bandwidth, not HBM. mode: 0=pre_weights,
// 1=negative loss partials, 2=positive loss partials.
// ---------------------------------------------------------------------------
__global__ __launch_bounds__(256) void edge_kernel(
    const float* __restrict__ latL, const float* __restrict__ latA,
    const float* __restrict__ aa,   const float* __restrict__ fd,
    const int* __restrict__ ei, const int* __restrict__ ej, int nE,
    const float* __restrict__ ewp, const float* __restrict__ swp,
    int mode, float* __restrict__ outPw, float* __restrict__ partial)
{
    __shared__ float sred[8];
    const int lane = threadIdx.x & 31;
    const int wave = threadIdx.x >> 5;
    const int wid  = blockIdx.x * 8 + wave;
    const int nw   = gridDim.x * 8;
    const float ew = *ewp;
    const float sw = *swp;

    float accum = 0.0f;
    for (int e = wid; e < nE; e += nw) {
        const int i = ei[e];
        const int j = ej[e];
        float4 a0 = ((const float4*)(latL + (size_t)i * HID))[lane];
        float4 a1 = ((const float4*)(latL + (size_t)j * HID))[lane];
        float4 c0 = ((const float4*)(latA + (size_t)i * HID))[lane];
        float4 c1 = ((const float4*)(latA + (size_t)j * HID))[lane];
        float s1 = a0.x * a1.x + a0.y * a1.y + a0.z * a1.z + a0.w * a1.w;
        float s2 = c0.x * c1.x + c0.y * c1.y + c0.z * c1.z + c0.w * c1.w;
#pragma unroll
        for (int off = 16; off > 0; off >>= 1) {
            s1 += __shfl_xor(s1, off, 32);
            s2 += __shfl_xor(s2, off, 32);
        }
        const size_t ij = (size_t)i * N_NODES + j;
        const float lc = sigmoidf(s1);
        const float ac = sigmoidf(s2 * aa[ij]);
        const float s  = sigmoidf(ew * lc + sw * ac);

        if (mode == 0) {
            float pw = fmaxf(s, 0.0f);
            if (pw < T_DELETE) pw = 0.0f;
            if (lane == 0) outPw[e] = pw;
        } else {
            const float f = fd[ij] * SIGMA_INV;
            const float q = f * f;
            const float w = (mode == 1) ? expf(q) : expf(-q);
            const float d = (mode == 1) ? s : (s - 1.0f);
            accum += w * d * d;
        }
    }

    if (mode != 0) {
        if (lane == 0) sred[wave] = accum;
        __syncthreads();
        if (threadIdx.x == 0) {
            float t = 0.0f;
#pragma unroll
            for (int k = 0; k < 8; ++k) t += sred[k];
            partial[blockIdx.x] = t;   // deterministic: fixed edge->block mapping
        }
    }
}

__global__ void finalize_kernel(const float* __restrict__ partial, int nPart,
                                float scale, float* __restrict__ out)
{
    if (threadIdx.x == 0 && blockIdx.x == 0) {
        float t = 0.0f;
        for (int k = 0; k < nPart; ++k) t += partial[k];
        out[0] = t * scale;
    }
}

// ---------------------------------------------------------------------------
extern "C" void kernel_launch(void* const* d_in, const int* in_sizes, int n_in,
                              void* d_out, int out_size, void* d_ws, size_t ws_size,
                              hipStream_t stream) {
    const float* emb = (const float*)d_in[0];
    const float* aa  = (const float*)d_in[1];
    const float* fd  = (const float*)d_in[2];
    const float* W1L = (const float*)d_in[3];
    const float* b1L = (const float*)d_in[4];
    const float* W2L = (const float*)d_in[5];
    const float* b2L = (const float*)d_in[6];
    const float* W1A = (const float*)d_in[7];
    const float* b1A = (const float*)d_in[8];
    const float* W2A = (const float*)d_in[9];
    const float* b2A = (const float*)d_in[10];
    const float* ewp = (const float*)d_in[11];
    const float* swp = (const float*)d_in[12];
    const int* epos = (const int*)d_in[13]; const int nPos = in_sizes[13] / 2;
    const int* eneg = (const int*)d_in[14]; const int nNeg = in_sizes[14] / 2;
    const int* epot = (const int*)d_in[15]; const int nPot = in_sizes[15] / 2;

    float* out = (float*)d_out;
    float* ws  = (float*)d_ws;

    // ws layout (floats):
    float* latL    = ws;                                   // N*HID
    float* latA    = ws + (size_t)N_NODES * HID;           // N*HID
    float* partPos = ws + 2 * (size_t)N_NODES * HID;       // 256
    float* partNeg = partPos + 256;                        // 256
    unsigned* bfArea = (unsigned*)(partNeg + 256);
    unsigned* embB = bfArea;                               // N*EMB/2 uints
    unsigned* W1LB = embB + (size_t)N_NODES * EMB / 2;
    unsigned* W2LB = W1LB + EMB * HID / 2;
    unsigned* W1AB = W2LB + HID * HID / 2;
    unsigned* W2AB = W1AB + EMB * HID / 2;

    // Stage 0: one-pass bf16 conversion of GEMM operands
    cvt_bf16_kernel<<<1024, 256, 0, stream>>>(emb, embB, N_NODES * EMB / 2);
    cvt_bf16_kernel<<<64, 256, 0, stream>>>(W1L, W1LB, EMB * HID / 2);
    cvt_bf16_kernel<<<32, 256, 0, stream>>>(W2L, W2LB, HID * HID / 2);
    cvt_bf16_kernel<<<64, 256, 0, stream>>>(W1A, W1AB, EMB * HID / 2);
    cvt_bf16_kernel<<<32, 256, 0, stream>>>(W2A, W2AB, HID * HID / 2);

    // Stage 1: latents for both branches (48 row-blocks x 2 branches)
    dim3 g1(N_NODES / 128, 2);
    mlp_wmma_kernel<<<g1, 256, 0, stream>>>(
        (const unsigned short*)embB,
        (const unsigned short*)W1LB, b1L, (const unsigned short*)W2LB, b2L,
        (const unsigned short*)W1AB, b1A, (const unsigned short*)W2AB, b2A,
        latL, latA);

    // Stage 2: potential-edge pre_weights straight into out[1..]
    edge_kernel<<<2048, 256, 0, stream>>>(latL, latA, aa, fd,
                                          epot, epot + nPot, nPot,
                                          ewp, swp, 0, out + 1, nullptr);
    // negative / positive loss partials (deterministic block reductions)
    edge_kernel<<<256, 256, 0, stream>>>(latL, latA, aa, fd,
                                         eneg, eneg + nNeg, nNeg,
                                         ewp, swp, 1, nullptr, partNeg);
    edge_kernel<<<256, 256, 0, stream>>>(latL, latA, aa, fd,
                                         epos, epos + nPos, nPos,
                                         ewp, swp, 2, nullptr, partPos);

    const float scale = (float)N_NODES / (float)(nNeg + nPos);
    finalize_kernel<<<1, 32, 0, stream>>>(partPos, 512, scale, out);
}
